// ParallelLinear_35553739276693
// MI455X (gfx1250) — matrix operand order repaired and checked by reading, not compiled, not run on hardware
//
#include <hip/hip_runtime.h>
#include <cstdint>
#include <cstddef>

#define LNUM 4
#define DIN  2048
#define DOUT 2048
#define BTOK 16384
#define KC   32
#define NKC  (DIN / KC)   // 64

typedef __attribute__((ext_vector_type(16))) __bf16 v16bf;
typedef __attribute__((ext_vector_type(8)))  float  v8f;

// ---- workspace layout (bytes) ----
static constexpr size_t OFF_COUNTS = 0;                         // 4 ints
static constexpr size_t OFF_GATHER = 1024;                      // L*BTOK ints = 256KB
static constexpr size_t OFF_XHI    = 1u << 20;                  // 1MB aligned start
static constexpr size_t SZ_X       = (size_t)BTOK * DIN * 2;    // 64MB (bf16)
static constexpr size_t OFF_XLO    = OFF_XHI + SZ_X;
static constexpr size_t SZ_W       = (size_t)LNUM * DOUT * DIN * 2; // 32MB (bf16)
static constexpr size_t OFF_WHI    = OFF_XLO + SZ_X;
static constexpr size_t OFF_WLO    = OFF_WHI + SZ_W;

__device__ __forceinline__ unsigned short f2bf(float f) {
  unsigned u = __builtin_bit_cast(unsigned, f);
  u += 0x7FFFu + ((u >> 16) & 1u);           // round-to-nearest-even
  return (unsigned short)(u >> 16);
}
__device__ __forceinline__ float bf2f(unsigned short h) {
  unsigned u = ((unsigned)h) << 16;
  return __builtin_bit_cast(float, u);
}

// CDNA5 async global->LDS DMA (16B per lane), tracked by ASYNCcnt.
__device__ __forceinline__ void async_ld16(unsigned lds_off, const void* g) {
  asm volatile("global_load_async_to_lds_b128 %0, %1, off"
               :: "v"(lds_off), "v"((unsigned long long)(uintptr_t)g)
               : "memory");
}

// ---- kernel 1: bucket tokens by expert ----
__global__ void bin_kernel(const int* __restrict__ lidx,
                           int* __restrict__ counts,
                           int* __restrict__ gather) {
  int t = blockIdx.x * blockDim.x + threadIdx.x;
  if (t < BTOK) {
    int e = lidx[t] & (LNUM - 1);
    int slot = atomicAdd(&counts[e], 1);
    gather[e * BTOK + slot] = t;
  }
}

// ---- kernel 2: x fp32 -> bf16 hi/lo split ----
__global__ void cvt_x_kernel(const float* __restrict__ x,
                             unsigned short* __restrict__ xhi,
                             unsigned short* __restrict__ xlo) {
  size_t i = ((size_t)blockIdx.x * blockDim.x + threadIdx.x) * 4;
  float4 v = *(const float4*)(x + i);
  unsigned short h0 = f2bf(v.x), h1 = f2bf(v.y), h2 = f2bf(v.z), h3 = f2bf(v.w);
  unsigned short l0 = f2bf(v.x - bf2f(h0));
  unsigned short l1 = f2bf(v.y - bf2f(h1));
  unsigned short l2 = f2bf(v.z - bf2f(h2));
  unsigned short l3 = f2bf(v.w - bf2f(h3));
  uint2 hh, ll;
  hh.x = (unsigned)h0 | ((unsigned)h1 << 16);
  hh.y = (unsigned)h2 | ((unsigned)h3 << 16);
  ll.x = (unsigned)l0 | ((unsigned)l1 << 16);
  ll.y = (unsigned)l2 | ((unsigned)l3 << 16);
  *(uint2*)(xhi + i) = hh;
  *(uint2*)(xlo + i) = ll;
}

// ---- kernel 3: W [L][K][N] fp32 -> Wt [L][N][K] bf16 hi/lo (transposed) ----
__global__ void cvt_w_kernel(const float* __restrict__ W,
                             unsigned short* __restrict__ whi,
                             unsigned short* __restrict__ wlo) {
  __shared__ float tile[32][33];
  int e = blockIdx.z;
  int n0 = blockIdx.x * 32, k0 = blockIdx.y * 32;
  int tx = threadIdx.x, ty = threadIdx.y;     // 32 x 8
  const float* We = W + (size_t)e * DIN * DOUT;
#pragma unroll
  for (int rr = 0; rr < 4; ++rr) {
    int k = k0 + ty + rr * 8;
    tile[ty + rr * 8][tx] = We[(size_t)k * DOUT + (n0 + tx)];
  }
  __syncthreads();
  unsigned short* whe = whi + (size_t)e * DOUT * DIN;
  unsigned short* wle = wlo + (size_t)e * DOUT * DIN;
#pragma unroll
  for (int rr = 0; rr < 4; ++rr) {
    int n = n0 + ty + rr * 8;
    float v = tile[tx][ty + rr * 8];
    unsigned short h = f2bf(v);
    unsigned short l = f2bf(v - bf2f(h));
    whe[(size_t)n * DIN + (k0 + tx)] = h;
    wle[(size_t)n * DIN + (k0 + tx)] = l;
  }
}

// ---- kernel 4: grouped GEMM, bf16 WMMA hi/lo split, async double-buffered LDS ----
// Block tile 64 tokens x 128 outputs; 8 waves = 4(M) x 2(N); wave tile 16x64.
#define LPAD 48   // LDS row stride in bf16 elems: 96B (16B-aligned, bank-spreading)

__global__ __launch_bounds__(256) void moe_gemm_kernel(
    const int* __restrict__ counts, const int* __restrict__ gather,
    const unsigned short* __restrict__ xhi, const unsigned short* __restrict__ xlo,
    const unsigned short* __restrict__ whi, const unsigned short* __restrict__ wlo,
    const float* __restrict__ bias, float* __restrict__ out) {
  __shared__ int toks[64];
  __shared__ __align__(16) unsigned short sAh[2][64][LPAD];
  __shared__ __align__(16) unsigned short sAl[2][64][LPAD];
  __shared__ __align__(16) unsigned short sBh[2][128][LPAD];
  __shared__ __align__(16) unsigned short sBl[2][128][LPAD];

  int slot  = blockIdx.x;                  // 0 .. L*(BTOK/64)-1
  int e     = slot >> 8;                   // BTOK/64 = 256 tiles per expert
  int base  = (slot & 255) * 64;
  int cnt   = counts[e];
  if (base >= cnt) return;                 // uniform early exit
  int nBase = blockIdx.y * 128;
  int tid   = threadIdx.x;

  const int* ge = gather + e * BTOK;
  if (tid < 64) {
    int g = base + tid;
    toks[tid] = (g < cnt) ? ge[g] : ge[base];   // pad with valid token; masked at store
  }
  __syncthreads();

  int lane  = tid & 31;
  int w     = tid >> 5;
  int waveM = w & 3;                       // 4 M-groups of 16 rows
  int waveN = w >> 2;                      // 2 N-groups of 64 cols
  int hsel  = lane >> 4;
  int l16   = lane & 15;

  // staging assignments: A = 1x16B per array per thread, B = 2x16B per array
  int rA = tid >> 2, segA = (tid & 3) * 8;      // 64 rows x 4 segs of 8 elems
  int cB = tid >> 1, segB = (tid & 1) * 16;     // 128 cols x 2 segs of 16 elems
  const unsigned short* gAh = xhi + (size_t)toks[rA] * DIN + segA;
  const unsigned short* gAl = xlo + (size_t)toks[rA] * DIN + segA;
  const unsigned short* gBh = whi + (size_t)e * DOUT * DIN + (size_t)(nBase + cB) * DIN + segB;
  const unsigned short* gBl = wlo + (size_t)e * DOUT * DIN + (size_t)(nBase + cB) * DIN + segB;
  unsigned dAh[2] = { (unsigned)(uintptr_t)&sAh[0][rA][segA],
                      (unsigned)(uintptr_t)&sAh[1][rA][segA] };
  unsigned dAl[2] = { (unsigned)(uintptr_t)&sAl[0][rA][segA],
                      (unsigned)(uintptr_t)&sAl[1][rA][segA] };
  unsigned dBh[2] = { (unsigned)(uintptr_t)&sBh[0][cB][segB],
                      (unsigned)(uintptr_t)&sBh[1][cB][segB] };
  unsigned dBl[2] = { (unsigned)(uintptr_t)&sBl[0][cB][segB],
                      (unsigned)(uintptr_t)&sBl[1][cB][segB] };

  auto stage = [&](int b, int k0) {        // 6 async DMAs per thread (per wave: cnt=6)
    async_ld16(dAh[b],      gAh + k0);
    async_ld16(dAl[b],      gAl + k0);
    async_ld16(dBh[b],      gBh + k0);
    async_ld16(dBh[b] + 16, gBh + k0 + 8);
    async_ld16(dBl[b],      gBl + k0);
    async_ld16(dBl[b] + 16, gBl + k0 + 8);
  };

  v8f acc0 = {0.f,0.f,0.f,0.f,0.f,0.f,0.f,0.f};
  v8f acc1 = acc0, acc2 = acc0, acc3 = acc0;

  stage(0, 0);
  for (int kc = 0; kc < NKC; ++kc) {
    if (kc + 1 < NKC) {
      stage((kc + 1) & 1, (kc + 1) * KC);
      // 6 just issued; async loads complete in order, so <=6 means prev batch landed
      asm volatile("s_wait_asynccnt 0x6" ::: "memory");
    } else {
      asm volatile("s_wait_asynccnt 0x0" ::: "memory");
    }
    __syncthreads();

    int b = kc & 1;
    int arow = waveM * 16 + l16;
    // A frags (contiguous, matching B): lane<16: K0-15; lane>=16: K16-31
    union { struct { uint4 lo; uint4 hi; } q; v16bf v; } ah, al;
    ah.q.lo = *(const uint4*)&sAh[b][arow][hsel * 16];
    ah.q.hi = *(const uint4*)&sAh[b][arow][hsel * 16 + 8];
    al.q.lo = *(const uint4*)&sAl[b][arow][hsel * 16];
    al.q.hi = *(const uint4*)&sAl[b][arow][hsel * 16 + 8];

#pragma unroll
    for (int ns = 0; ns < 4; ++ns) {
      int c = waveN * 64 + ns * 16 + l16;  // B layout: lane<16 K0-15, lane>=16 K16-31
      union { struct { uint4 lo; uint4 hi; } q; v16bf v; } bh, bl;
      bh.q.lo = *(const uint4*)&sBh[b][c][hsel * 16];
      bh.q.hi = *(const uint4*)&sBh[b][c][hsel * 16 + 8];
      bl.q.lo = *(const uint4*)&sBl[b][c][hsel * 16];
      bl.q.hi = *(const uint4*)&sBl[b][c][hsel * 16 + 8];

      v8f& acc = (ns == 0) ? acc0 : (ns == 1) ? acc1 : (ns == 2) ? acc2 : acc3;
      acc = __builtin_amdgcn_wmma_f32_16x16x32_bf16(false, ah.v, false, bh.v,
                                                    (short)0, acc, false, false);
      acc = __builtin_amdgcn_wmma_f32_16x16x32_bf16(false, ah.v, false, bl.v,
                                                    (short)0, acc, false, false);
      acc = __builtin_amdgcn_wmma_f32_16x16x32_bf16(false, al.v, false, bh.v,
                                                    (short)0, acc, false, false);
    }
    __syncthreads();
  }

  // epilogue: bias + predicated scatter (C layout: VGPR i -> M=i / M=8+i by lane half)
  const float* be = bias + e * DOUT;
#pragma unroll
  for (int ns = 0; ns < 4; ++ns) {
    v8f acc = (ns == 0) ? acc0 : (ns == 1) ? acc1 : (ns == 2) ? acc2 : acc3;
    int ncol = nBase + waveN * 64 + ns * 16 + l16;
    float bv = be[ncol];
#pragma unroll
    for (int i = 0; i < 8; ++i) {
      int M = waveM * 16 + i + hsel * 8;
      if (base + M < cnt) {
        out[(size_t)toks[M] * DOUT + ncol] = acc[i] + bv;
      }
    }
  }
}

extern "C" void kernel_launch(void* const* d_in, const int* in_sizes, int n_in,
                              void* d_out, int out_size, void* d_ws, size_t ws_size,
                              hipStream_t stream) {
  (void)in_sizes; (void)n_in; (void)out_size; (void)ws_size;
  const float* x    = (const float*)d_in[0];
  const int*   lidx = (const int*)d_in[1];
  const float* W    = (const float*)d_in[2];
  const float* bias = (const float*)d_in[3];
  float*       out  = (float*)d_out;

  char* ws = (char*)d_ws;
  int* counts = (int*)(ws + OFF_COUNTS);
  int* gather = (int*)(ws + OFF_GATHER);
  unsigned short* xhi = (unsigned short*)(ws + OFF_XHI);
  unsigned short* xlo = (unsigned short*)(ws + OFF_XLO);
  unsigned short* whi = (unsigned short*)(ws + OFF_WHI);
  unsigned short* wlo = (unsigned short*)(ws + OFF_WLO);

  hipMemsetAsync(counts, 0, 64, stream);
  bin_kernel<<<BTOK / 256, 256, 0, stream>>>(lidx, counts, gather);
  cvt_x_kernel<<<(unsigned)((size_t)BTOK * DIN / 4 / 256), 256, 0, stream>>>(x, xhi, xlo);
  cvt_w_kernel<<<dim3(DOUT / 32, DIN / 32, LNUM), dim3(32, 8), 0, stream>>>(W, whi, wlo);
  moe_gemm_kernel<<<dim3(LNUM * (BTOK / 64), DOUT / 128), 256, 0, stream>>>(
      counts, gather, xhi, xlo, whi, wlo, bias, out);
}
